// GAT_n2v_mean_68968584839882
// MI455X (gfx1250) — compile-verified, matrix operand order checked by reading, not run on hardware
//
#include <hip/hip_runtime.h>
#include <hip/hip_bf16.h>

#define N_NODES 50000
#define N_EDGES 800000
#define TOT_EDGES (N_EDGES + N_NODES)
#define N_GRAPHS 64
#define NHEADS 4

typedef __attribute__((ext_vector_type(16))) __bf16 v16bf;
typedef __attribute__((ext_vector_type(8)))  __bf16 v8bf;
typedef __attribute__((ext_vector_type(8)))  float  v8f;

// ---------- order-preserving float<->uint for atomicMax on floats ----------
__device__ __forceinline__ unsigned enc_f32(float f) {
  unsigned u = __float_as_uint(f);
  return (u & 0x80000000u) ? ~u : (u | 0x80000000u);
}
__device__ __forceinline__ float dec_f32(unsigned u) {
  unsigned b = (u & 0x80000000u) ? (u & 0x7fffffffu) : ~u;
  return __uint_as_float(b);
}

// ---------- misc small kernels ----------
__global__ void k_f32_to_bf16(const float* __restrict__ src, __bf16* __restrict__ dst, int n) {
  int i = blockIdx.x * blockDim.x + threadIdx.x;
  if (i < n) dst[i] = (__bf16)src[i];
}

// W [K x Nout] f32 -> Wt [Nout x K] bf16
__global__ void k_transpose_bf16(const float* __restrict__ W, __bf16* __restrict__ Wt,
                                 int K, int Nout) {
  int i = blockIdx.x * blockDim.x + threadIdx.x;
  if (i >= K * Nout) return;
  int n = i % Nout, k = i / Nout;
  Wt[(size_t)n * K + k] = (__bf16)W[(size_t)k * Nout + n];
}

__device__ __forceinline__ v16bf frag_ld(const __bf16* p) {
  v8bf lo = *(const v8bf*)(p);
  v8bf hi = *(const v8bf*)(p + 16);
  return __builtin_shufflevector(lo, hi, 0,1,2,3,4,5,6,7,8,9,10,11,12,13,14,15);
}

// ---------- WMMA GEMM: C[M x Nout] = A[M x K](bf16) * Bt[Nout x K](bf16)^T ----------
// block = 128 threads = 4 waves. Each wave computes a 16(M) x 64(N) strip:
// one A fragment is reused across 4 B fragments -> 4 v_wmma per A load.
// All 4 waves in the block share the same M-tile, so A loads hit L0.
// 16-bit A/B fragment layout: lanes 0-15 hold row=lane, K = k..k+7 / k+16..k+23;
// lanes 16-31 hold row=lane-16, K = k+8..k+15 / k+24..k+31 -> two 16B loads per lane.
__global__ void k_gemm_wmma_bf16(const __bf16* __restrict__ A, const __bf16* __restrict__ Bt,
                                 float* __restrict__ C, int M, int K, int Nout) {
  int lane = threadIdx.x & 31;
  int wave = threadIdx.x >> 5;
  int colbase = (blockIdx.x * 4 + wave) * 64;   // 64 output columns per wave
  int tileM = blockIdx.y;
  int r15 = lane & 15;
  int khi = (lane >> 4) << 3;                   // 0 or 8
  const __bf16* arow = A  + (size_t)(tileM * 16 + r15) * K + khi;
  const __bf16* brow = Bt + (size_t)(colbase + r15) * K + khi;
  const size_t bstride = (size_t)16 * K;        // 16 columns ahead in Bt
  v8f acc0 = {}, acc1 = {}, acc2 = {}, acc3 = {};
  for (int k = 0; k < K; k += 32) {
    __builtin_prefetch(arow + k + 256, 0, 0);   // global_prefetch_b8 down the A stream
    v16bf av  = frag_ld(arow + k);
    v16bf bv0 = frag_ld(brow + k);
    v16bf bv1 = frag_ld(brow + k + bstride);
    v16bf bv2 = frag_ld(brow + k + 2 * bstride);
    v16bf bv3 = frag_ld(brow + k + 3 * bstride);
    acc0 = __builtin_amdgcn_wmma_f32_16x16x32_bf16(false, av, false, bv0, (short)0, acc0, false, false);
    acc1 = __builtin_amdgcn_wmma_f32_16x16x32_bf16(false, av, false, bv1, (short)0, acc1, false, false);
    acc2 = __builtin_amdgcn_wmma_f32_16x16x32_bf16(false, av, false, bv2, (short)0, acc2, false, false);
    acc3 = __builtin_amdgcn_wmma_f32_16x16x32_bf16(false, av, false, bv3, (short)0, acc3, false, false);
  }
  // C/D layout: VGPR r -> M = r (lanes 0-15) or r+8 (lanes 16-31), N = lane&15
  int mbase = tileM * 16 + khi;
  int nb = colbase + r15;
#pragma unroll
  for (int r = 0; r < 8; ++r) {
    float* crow = C + (size_t)(mbase + r) * Nout + nb;
    crow[0]  = acc0[r];
    crow[16] = acc1[r];
    crow[32] = acc2[r];
    crow[48] = acc3[r];
  }
}

// ---------- degree + per-dst mean edge feature (self-loop fill_value='mean') ----------
__global__ void k_deg_emean(const int* __restrict__ ei, const float* __restrict__ ef,
                            float* __restrict__ deg, float* __restrict__ emean) {
  int e = blockIdx.x * blockDim.x + threadIdx.x;
  if (e >= N_EDGES) return;
  int dst = ei[N_EDGES + e];
  atomicAdd(&deg[dst], 1.0f);
#pragma unroll
  for (int k = 0; k < 6; ++k) atomicAdd(&emean[dst * 6 + k], ef[(size_t)e * 6 + k]);
}

__global__ void k_emean_fin(float* __restrict__ emean, const float* __restrict__ deg) {
  int i = blockIdx.x * blockDim.x + threadIdx.x;
  if (i >= N_NODES) return;
  float d = 1.0f / fmaxf(deg[i], 1.0f);
#pragma unroll
  for (int k = 0; k < 6; ++k) emean[i * 6 + k] *= d;
}

// ---------- ve[k][h] = sum_c We[k, h*C+c] * ae[h, c] ----------
__global__ void k_ve(const float* __restrict__ We, const float* __restrict__ ae,
                     float* __restrict__ ve, int C) {
  int t = threadIdx.x;
  if (t >= 6 * NHEADS) return;
  int k = t / NHEADS, h = t % NHEADS;
  int HC = NHEADS * C;
  float acc = 0.f;
  for (int c = 0; c < C; ++c) acc += We[k * HC + h * C + c] * ae[h * C + c];
  ve[k * NHEADS + h] = acc;
}

// ---------- per-node attention logit halves ----------
__global__ void k_al_nodes(const float* __restrict__ xs, const float* __restrict__ as_,
                           const float* __restrict__ ad_, float* __restrict__ als,
                           float* __restrict__ ald, int C) {
  int t = blockIdx.x * blockDim.x + threadIdx.x;
  if (t >= N_NODES * NHEADS) return;
  int i = t / NHEADS, h = t % NHEADS;
  const float* xr = xs + (size_t)i * NHEADS * C + h * C;
  float s = 0.f, d = 0.f;
  for (int c = 0; c < C; ++c) { float v = xr[c]; s += v * as_[h * C + c]; d += v * ad_[h * C + c]; }
  als[t] = s; ald[t] = d;
}

// ---------- edge logits + segment max ----------
__global__ void k_edge_logits(const int* __restrict__ ei, const float* __restrict__ ef,
                              const float* __restrict__ emean, const float* __restrict__ ve,
                              const float* __restrict__ als, const float* __restrict__ ald,
                              float* __restrict__ al, unsigned* __restrict__ m) {
  int e = blockIdx.x * blockDim.x + threadIdx.x;
  if (e >= TOT_EDGES) return;
  int src, dst; const float* ea;
  if (e < N_EDGES) { src = ei[e]; dst = ei[N_EDGES + e]; ea = ef + (size_t)e * 6; }
  else            { src = dst = e - N_EDGES;             ea = emean + (size_t)(e - N_EDGES) * 6; }
  float eav[6];
#pragma unroll
  for (int k = 0; k < 6; ++k) eav[k] = ea[k];
#pragma unroll
  for (int h = 0; h < NHEADS; ++h) {
    float ale = 0.f;
#pragma unroll
    for (int k = 0; k < 6; ++k) ale += eav[k] * ve[k * NHEADS + h];
    float v = als[src * NHEADS + h] + ald[dst * NHEADS + h] + ale;
    v = (v > 0.f) ? v : 0.2f * v;                 // leaky_relu(0.2)
    al[(size_t)e * NHEADS + h] = v;
    atomicMax(&m[dst * NHEADS + h], enc_f32(v));
  }
}

// ---------- softmax numerator + segment denominator ----------
__global__ void k_edge_w(const int* __restrict__ ei, const float* __restrict__ al,
                         const unsigned* __restrict__ m, float* __restrict__ w,
                         float* __restrict__ denom) {
  int e = blockIdx.x * blockDim.x + threadIdx.x;
  if (e >= TOT_EDGES) return;
  int dst = (e < N_EDGES) ? ei[N_EDGES + e] : e - N_EDGES;
#pragma unroll
  for (int h = 0; h < NHEADS; ++h) {
    float mv = dec_f32(m[dst * NHEADS + h]);
    float x = __expf(al[(size_t)e * NHEADS + h] - mv);
    w[(size_t)e * NHEADS + h] = x;
    atomicAdd(&denom[dst * NHEADS + h], x);
  }
}

// ---------- normalize softmax weights once per (edge, head) ----------
__global__ void k_norm_w(const int* __restrict__ ei, float* __restrict__ w,
                         const float* __restrict__ denom) {
  int t = blockIdx.x * blockDim.x + threadIdx.x;
  if (t >= TOT_EDGES * NHEADS) return;
  int e = t / NHEADS, h = t % NHEADS;
  int dst = (e < N_EDGES) ? ei[N_EDGES + e] : e - N_EDGES;
  w[t] = w[t] / (denom[dst * NHEADS + h] + 1e-16f);
}

// ---------- weighted scatter aggregation: out[dst] += xs[src] * w_norm ----------
// one thread per (edge, 4-channel quad); gathers hit L2 (102MB working set < 192MB L2)
__global__ void k_aggregate(const int* __restrict__ ei, const float* __restrict__ xs,
                            const float* __restrict__ w, float* __restrict__ out, int C) {
  int HCq = NHEADS * C / 4;
  long long t = (long long)blockIdx.x * blockDim.x + threadIdx.x;
  if (t >= (long long)TOT_EDGES * HCq) return;
  int e  = (int)(t / HCq);
  int c4 = (int)(t % HCq) * 4;
  int h  = c4 / C;
  int src, dst;
  if (e < N_EDGES) { src = ei[e]; dst = ei[N_EDGES + e]; } else { src = dst = e - N_EDGES; }
  float scale = w[(size_t)e * NHEADS + h];
  const float4 v = *(const float4*)(xs + (size_t)src * NHEADS * C + c4);
  float* o = out + (size_t)dst * NHEADS * C + c4;
  atomicAdd(o + 0, v.x * scale);
  atomicAdd(o + 1, v.y * scale);
  atomicAdd(o + 2, v.z * scale);
  atomicAdd(o + 3, v.w * scale);
}

// ---------- bias + BN(eval) + ELU, emit f32 (pooling) + bf16 (next GEMM) ----------
__global__ void k_post(const float* __restrict__ acc, const float* __restrict__ b,
                       const float* __restrict__ g, const float* __restrict__ bb,
                       float* __restrict__ hf, __bf16* __restrict__ hb, int HC) {
  long long t = (long long)blockIdx.x * blockDim.x + threadIdx.x;
  if (t >= (long long)N_NODES * HC) return;
  int j = (int)(t % HC);
  float inv = rsqrtf(1.0f + 1e-5f);
  float v = acc[t] + b[j];
  v = g[j] * v * inv + bb[j];
  v = (v > 0.f) ? v : (__expf(v) - 1.f);          // ELU
  hf[t] = v;
  hb[t] = (__bf16)v;
}

// ---------- pooling ----------
__global__ void k_pool_cnt(const int* __restrict__ batch, float* __restrict__ cnt) {
  int i = blockIdx.x * blockDim.x + threadIdx.x;
  if (i < N_NODES) atomicAdd(&cnt[batch[i]], 1.0f);
}
__global__ void k_pool_sum(const int* __restrict__ batch, const float* __restrict__ h,
                           float* __restrict__ pooled) {
  long long t = (long long)blockIdx.x * blockDim.x + threadIdx.x;
  if (t >= (long long)N_NODES * 256) return;
  int i = (int)(t / 256), j = (int)(t % 256);
  atomicAdd(&pooled[batch[i] * 256 + j], h[t]);
}

// ---------- head: mean -> FC1 -> BN -> ELU -> FC2 -> log_softmax ----------
__global__ void k_head(const float* __restrict__ pooled, const float* __restrict__ cnt,
                       const float* __restrict__ Wf1, const float* __restrict__ bf1,
                       const float* __restrict__ gf, const float* __restrict__ bbf,
                       const float* __restrict__ Wf2, const float* __restrict__ bf2,
                       float* __restrict__ outp) {
  int g = blockIdx.x;      // graph
  int t = threadIdx.x;     // 0..31 -> one FC1 output each
  __shared__ float sh[32];
  __shared__ float sl[2];
  float invc = 1.0f / fmaxf(cnt[g], 1.0f);
  float acc = bf1[t];
  for (int j = 0; j < 256; ++j) acc += pooled[g * 256 + j] * invc * Wf1[j * 32 + t];
  float inv = rsqrtf(1.0f + 1e-5f);
  acc = gf[t] * acc * inv + bbf[t];
  acc = (acc > 0.f) ? acc : (__expf(acc) - 1.f);
  sh[t] = acc;
  __syncthreads();
  if (t < 2) {
    float l = bf2[t];
    for (int j = 0; j < 32; ++j) l += sh[j] * Wf2[j * 2 + t];
    sl[t] = l;
  }
  __syncthreads();
  if (t == 0) {
    float mx = fmaxf(sl[0], sl[1]);
    float lse = mx + logf(__expf(sl[0] - mx) + __expf(sl[1] - mx));
    outp[g * 2 + 0] = sl[0] - lse;
    outp[g * 2 + 1] = sl[1] - lse;
  }
}

// ---------- workspace layout (bytes) ----------
#define XS_OFF    0ull                      // N*512 f32      = 102,400,000
#define OUT_OFF   102400000ull              // N*512 f32
#define HA_OFF    204800000ull              // N*512 f32
#define HBF_OFF   307200000ull              // N*512 bf16     =  51,200,000
#define WT_OFF    358400000ull              // 512*512 bf16 max
#define VE_OFF    358924544ull              // 24 f32
#define DEG_OFF   358924800ull              // N f32
#define EMEAN_OFF 359125248ull              // N*6 f32
#define AL_OFF    360325376ull              // TOT*4 f32      =  13,600,000
#define WBUF_OFF  373925376ull              // TOT*4 f32
#define M_OFF     387525376ull              // N*4 u32
#define DEN_OFF   388325376ull              // N*4 f32
#define ALS_OFF   389125376ull              // N*4 f32
#define ALD_OFF   389925376ull              // N*4 f32
#define POOL_OFF  390725376ull              // 64*256 f32
#define CNT_OFF   390790912ull              // 64 f32

extern "C" void kernel_launch(void* const* d_in, const int* in_sizes, int n_in,
                              void* d_out, int out_size, void* d_ws, size_t ws_size,
                              hipStream_t stream) {
  const float* x    = (const float*)d_in[0];
  const float* ef   = (const float*)d_in[1];
  const float* Wf1  = (const float*)d_in[26];
  const float* bf1  = (const float*)d_in[27];
  const float* gf   = (const float*)d_in[28];
  const float* bbf  = (const float*)d_in[29];
  const float* Wf2  = (const float*)d_in[30];
  const float* bf2  = (const float*)d_in[31];
  const int*   ei   = (const int*)d_in[32];
  const int*   batch= (const int*)d_in[33];

  char* ws = (char*)d_ws;
  float*    xs    = (float*)(ws + XS_OFF);
  float*    outb  = (float*)(ws + OUT_OFF);
  float*    hA    = (float*)(ws + HA_OFF);
  __bf16*   hbf   = (__bf16*)(ws + HBF_OFF);
  __bf16*   Wt    = (__bf16*)(ws + WT_OFF);
  float*    ve    = (float*)(ws + VE_OFF);
  float*    deg   = (float*)(ws + DEG_OFF);
  float*    emean = (float*)(ws + EMEAN_OFF);
  float*    al    = (float*)(ws + AL_OFF);
  float*    wbuf  = (float*)(ws + WBUF_OFF);
  unsigned* m     = (unsigned*)(ws + M_OFF);
  float*    denom = (float*)(ws + DEN_OFF);
  float*    als   = (float*)(ws + ALS_OFF);
  float*    ald   = (float*)(ws + ALD_OFF);
  float*    pooled= (float*)(ws + POOL_OFF);
  float*    cnt   = (float*)(ws + CNT_OFF);

  // ---- self-loop edge-attr mean ----
  hipMemsetAsync(deg, 0, N_NODES * 4, stream);
  hipMemsetAsync(emean, 0, N_NODES * 6 * 4, stream);
  k_deg_emean<<<(N_EDGES + 255) / 256, 256, 0, stream>>>(ei, ef, deg, emean);
  k_emean_fin<<<(N_NODES + 255) / 256, 256, 0, stream>>>(emean, deg);

  // ---- layer-1 input -> bf16 ----
  k_f32_to_bf16<<<(N_NODES * 32 + 255) / 256, 256, 0, stream>>>(x, hbf, N_NODES * 32);

  const int Kin[3] = {32, 256, 512};
  const int Cc[3]  = {64, 128, 64};
  for (int li = 0; li < 3; ++li) {
    int base = 2 + 8 * li;
    const float* W   = (const float*)d_in[base + 0];
    const float* We  = (const float*)d_in[base + 1];
    const float* as_ = (const float*)d_in[base + 2];
    const float* ad_ = (const float*)d_in[base + 3];
    const float* ae  = (const float*)d_in[base + 4];
    const float* b   = (const float*)d_in[base + 5];
    const float* g   = (const float*)d_in[base + 6];
    const float* bb  = (const float*)d_in[base + 7];
    int K = Kin[li], C = Cc[li], HC = NHEADS * C;

    k_transpose_bf16<<<(K * HC + 255) / 256, 256, 0, stream>>>(W, Wt, K, HC);
    dim3 gg(HC / 256, N_NODES / 16);      // 4 waves/block, 64 cols per wave
    k_gemm_wmma_bf16<<<gg, 128, 0, stream>>>(hbf, Wt, xs, N_NODES, K, HC);

    k_ve<<<1, 32, 0, stream>>>(We, ae, ve, C);
    k_al_nodes<<<(N_NODES * NHEADS + 255) / 256, 256, 0, stream>>>(xs, as_, ad_, als, ald, C);

    hipMemsetAsync(m, 0, N_NODES * NHEADS * 4, stream);        // enc-space -inf
    hipMemsetAsync(denom, 0, N_NODES * NHEADS * 4, stream);
    hipMemsetAsync(outb, 0, (size_t)N_NODES * HC * 4, stream);

    k_edge_logits<<<(TOT_EDGES + 255) / 256, 256, 0, stream>>>(ei, ef, emean, ve, als, ald, al, m);
    k_edge_w<<<(TOT_EDGES + 255) / 256, 256, 0, stream>>>(ei, al, m, wbuf, denom);
    k_norm_w<<<(TOT_EDGES * NHEADS + 255) / 256, 256, 0, stream>>>(ei, wbuf, denom);

    long long aggT = (long long)TOT_EDGES * (HC / 4);
    k_aggregate<<<(unsigned)((aggT + 255) / 256), 256, 0, stream>>>(ei, xs, wbuf, outb, C);

    long long postT = (long long)N_NODES * HC;
    k_post<<<(unsigned)((postT + 255) / 256), 256, 0, stream>>>(outb, b, g, bb, hA, hbf, HC);
  }

  // ---- mean pool + head ----
  hipMemsetAsync(pooled, 0, N_GRAPHS * 256 * 4, stream);
  hipMemsetAsync(cnt, 0, N_GRAPHS * 4, stream);
  k_pool_cnt<<<(N_NODES + 255) / 256, 256, 0, stream>>>(batch, cnt);
  k_pool_sum<<<(N_NODES * 256 + 255) / 256, 256, 0, stream>>>(batch, hA, pooled);
  k_head<<<N_GRAPHS, 32, 0, stream>>>(pooled, cnt, Wf1, bf1, gf, bbf, Wf2, bf2, (float*)d_out);
}
